// MambaCompatFallback_52518860095963
// MI455X (gfx1250) — compile-verified, matrix operand order checked by reading, not run on hardware
//
#include <hip/hip_runtime.h>

#define D_MODEL 1024
#define D_STATE 16
#define D_CONV 4
#define D_INNER 2048
#define DT_RANK 64
#define BATCH 4
#define SEQ 2048
#define MROWS (BATCH * SEQ)            // 8192
#define PROJW (DT_RANK + 2 * D_STATE)  // 96

typedef __attribute__((ext_vector_type(16))) __bf16 v16bf;
typedef __attribute__((ext_vector_type(8)))  float  v8f;
typedef __attribute__((ext_vector_type(2)))  float  f32x2;
typedef __attribute__((ext_vector_type(2)))  __bf16 bf16x2;

union FragBF { v16bf v; uint4 q[2]; };

// Pairwise f32->bf16: maps to v_cvt_pk_bf16_f32 (no v_mov_b16 reassembly)
__device__ __forceinline__ ushort4 cvt4_bf16(float4 f) {
    union { bf16x2 h2[2]; ushort4 u; } t;
    t.h2[0] = __builtin_convertvector((f32x2){f.x, f.y}, bf16x2);
    t.h2[1] = __builtin_convertvector((f32x2){f.z, f.w}, bf16x2);
    return t.u;
}

// ---------------- bf16 WMMA GEMM: C = A(fp32) * B(fp32), f32 accumulate ----------
// BM=128, BN=128, BK=32, 256 threads = 8 waves (4 rows x 2 cols), wave tile 32x64
// Double-buffered LDS + register staging of the next tile.
#define BM 128
#define BN 128
#define BK 32
#define ASTR 40   // padded LDS stride (bf16 units) to spread banks
#define BSTR 40

template <int EPILOGUE>  // 0 = plain store, 1 = +bias then softplus
__global__ __launch_bounds__(256) void gemm_bf16_wmma(
    const float* __restrict__ A, int lda,
    const float* __restrict__ Bm, int ldb,
    const float* __restrict__ bias,
    float* __restrict__ C, int ldc,
    int M, int N, int K)
{
    __shared__ __align__(16) unsigned short Alds[2][BM * ASTR];
    __shared__ __align__(16) unsigned short Blds[2][BN * BSTR];

    const int tid  = threadIdx.x;
    const int lane = tid & 31;            // wave32
    const int wave = tid >> 5;            // 0..7
    const int wr   = wave & 3;            // wave row 0..3 -> M offset 32*wr
    const int wc   = wave >> 2;           // wave col 0..1 -> N offset 64*wc
    const int blockM = blockIdx.y * BM;
    const int blockN = blockIdx.x * BN;
    const bool fullN = (blockN + BN) <= N;

    v8f acc[2][4];
#pragma unroll
    for (int i = 0; i < 2; ++i)
#pragma unroll
        for (int j = 0; j < 4; ++j) acc[i][j] = (v8f)0.0f;

    // A loader: float4/thread/pass, 4 passes cover 128x32
    const int a_row = tid >> 3;            // 0..31  (+32/pass)
    const int a_c4  = (tid & 7) << 2;      // 0..28
    // B loader: float4/thread/pass, 4 passes cover 32x128
    const int b_k   = tid >> 5;            // 0..7   (+8/pass)
    const int b_c4  = (tid & 31) << 2;     // 0..124

    float4 aReg[4], bReg[4];

    auto stage_global = [&](int k0) {
#pragma unroll
        for (int p = 0; p < 4; ++p) {
            int r = a_row + p * 32;
            aReg[p] = *reinterpret_cast<const float4*>(
                &A[(size_t)(blockM + r) * lda + k0 + a_c4]);
        }
        if (fullN) {
#pragma unroll
            for (int p = 0; p < 4; ++p) {
                int kk = b_k + p * 8;
                bReg[p] = *reinterpret_cast<const float4*>(
                    &Bm[(size_t)(k0 + kk) * ldb + blockN + b_c4]);
            }
        } else {
#pragma unroll
            for (int p = 0; p < 4; ++p) {
                int kk = b_k + p * 8;
                const float* row = &Bm[(size_t)(k0 + kk) * ldb];
                float4 v;
                v.x = (blockN + b_c4 + 0 < N) ? row[blockN + b_c4 + 0] : 0.0f;
                v.y = (blockN + b_c4 + 1 < N) ? row[blockN + b_c4 + 1] : 0.0f;
                v.z = (blockN + b_c4 + 2 < N) ? row[blockN + b_c4 + 2] : 0.0f;
                v.w = (blockN + b_c4 + 3 < N) ? row[blockN + b_c4 + 3] : 0.0f;
                bReg[p] = v;
            }
        }
    };

    auto stage_to_lds = [&](int buf) {
#pragma unroll
        for (int p = 0; p < 4; ++p) {
            int r = a_row + p * 32;
            *reinterpret_cast<ushort4*>(&Alds[buf][r * ASTR + a_c4]) = cvt4_bf16(aReg[p]);
        }
#pragma unroll
        for (int p = 0; p < 4; ++p) {
            int kk = b_k + p * 8;
            union { bf16x2 h2[2]; __bf16 h[4]; } t;
            t.h2[0] = __builtin_convertvector((f32x2){bReg[p].x, bReg[p].y}, bf16x2);
            t.h2[1] = __builtin_convertvector((f32x2){bReg[p].z, bReg[p].w}, bf16x2);
#pragma unroll
            for (int i = 0; i < 4; ++i)
                reinterpret_cast<__bf16*>(Blds[buf])[(b_c4 + i) * BSTR + kk] = t.h[i];
        }
    };

    // prologue: stage first tile
    stage_global(0);
    stage_to_lds(0);
    __syncthreads();

    const int lo   = (lane < 16) ? 0 : 8;   // CDNA5 16-bit A/B fragment K-halves
    const int rsel = lane & 15;
    int cur = 0;

    for (int k0 = 0; k0 < K; k0 += BK) {
        const bool hasNext = (k0 + BK) < K;
        if (hasNext) stage_global(k0 + BK);   // loads in flight during compute

        FragBF aF[2], bF[4];
#pragma unroll
        for (int tm = 0; tm < 2; ++tm) {
            int row = wr * 32 + tm * 16 + rsel;
            aF[tm].q[0] = *reinterpret_cast<const uint4*>(&Alds[cur][row * ASTR + lo]);
            aF[tm].q[1] = *reinterpret_cast<const uint4*>(&Alds[cur][row * ASTR + 16 + lo]);
        }
#pragma unroll
        for (int tn = 0; tn < 4; ++tn) {
            int col = wc * 64 + tn * 16 + rsel;
            bF[tn].q[0] = *reinterpret_cast<const uint4*>(&Blds[cur][col * BSTR + lo]);
            bF[tn].q[1] = *reinterpret_cast<const uint4*>(&Blds[cur][col * BSTR + 16 + lo]);
        }
#pragma unroll
        for (int tm = 0; tm < 2; ++tm)
#pragma unroll
            for (int tn = 0; tn < 4; ++tn)
                acc[tm][tn] = __builtin_amdgcn_wmma_f32_16x16x32_bf16(
                    false, aF[tm].v, false, bF[tn].v,
                    (short)0, acc[tm][tn], false, false);

        if (hasNext) {
            stage_to_lds(cur ^ 1);            // waits loads here, not earlier
            __syncthreads();
            cur ^= 1;
        }
    }

    // ---- epilogue + store (D layout: vgpr v, lane l -> M = v + (l>=16)*8, N = l&15) ----
#pragma unroll
    for (int tm = 0; tm < 2; ++tm) {
        int mbase = blockM + wr * 32 + tm * 16 + ((lane >= 16) ? 8 : 0);
#pragma unroll
        for (int tn = 0; tn < 4; ++tn) {
            int n = blockN + wc * 64 + tn * 16 + (lane & 15);
            if (n < N) {
#pragma unroll
                for (int v = 0; v < 8; ++v) {
                    float val = acc[tm][tn][v];
                    if (EPILOGUE == 1) {
                        val += bias[n];
                        val = (val > 20.0f) ? val : logf(1.0f + __expf(val));
                    }
                    C[(size_t)(mbase + v) * ldc + n] = val;
                }
            }
        }
    }
}

// ---------------- causal depthwise conv (k=4) + SiLU, 4 channels/thread ----------
__global__ __launch_bounds__(256) void conv_silu_kernel(
    const float* __restrict__ xz, const float* __restrict__ w,
    const float* __restrict__ b, float* __restrict__ out)
{
    int idx = (blockIdx.x * 256 + threadIdx.x) * 4;   // 4 consecutive channels
    int d  = idx & (D_INNER - 1);
    int t  = (idx / D_INNER) & (SEQ - 1);
    int bb = idx / (D_INNER * SEQ);

    float4 wch[4];
#pragma unroll
    for (int j = 0; j < 4; ++j)
        wch[j] = *reinterpret_cast<const float4*>(&w[(d + j) * D_CONV]);

    float4 acc = *reinterpret_cast<const float4*>(&b[d]);
#pragma unroll
    for (int k = 0; k < D_CONV; ++k) {
        int tt = t - (D_CONV - 1) + k;
        if (tt >= 0) {
            float4 xv = *reinterpret_cast<const float4*>(
                &xz[((size_t)(bb * SEQ + tt)) * (2 * D_INNER) + d]);
            acc.x += ((const float*)&wch[0])[k] * xv.x;
            acc.y += ((const float*)&wch[1])[k] * xv.y;
            acc.z += ((const float*)&wch[2])[k] * xv.z;
            acc.w += ((const float*)&wch[3])[k] * xv.w;
        }
    }
    float4 r;
    r.x = acc.x / (1.0f + __expf(-acc.x));
    r.y = acc.y / (1.0f + __expf(-acc.y));
    r.z = acc.z / (1.0f + __expf(-acc.z));
    r.w = acc.w / (1.0f + __expf(-acc.w));
    *reinterpret_cast<float4*>(&out[idx]) = r;
}

// ---------------- selective scan: one scalar recurrence per (b,d,s) lane ---------
// wave32 = 2 channels x 16 states; block = 16 channels; grid = (D_INNER/16, B)
__global__ __launch_bounds__(256) void scan_kernel(
    const float* __restrict__ dtm, const float* __restrict__ u,
    const float* __restrict__ proj, const float* __restrict__ xz,
    const float* __restrict__ A_log, const float* __restrict__ Dp,
    float* __restrict__ yact)
{
    const int tid = threadIdx.x;
    const int s   = tid & 15;
    const int d   = blockIdx.x * 16 + (tid >> 4);
    const int bb  = blockIdx.y;
    const float a  = -__expf(A_log[d * D_STATE + s]);
    const float dp = Dp[d];
    float state = 0.0f;
    const size_t baseRow = (size_t)bb * SEQ;
    for (int t = 0; t < SEQ; ++t) {
        size_t row = baseRow + t;
        float dt_v = dtm[row * D_INNER + d];              // broadcast over 16 lanes
        float u_v  = u[row * D_INNER + d];
        float bv   = proj[row * PROJW + DT_RANK + s];
        float cv   = proj[row * PROJW + DT_RANK + D_STATE + s];
        float da   = __expf(dt_v * a);
        state = __fmaf_rn(da, state, dt_v * bv * u_v);
        float y = state * cv;                              // reduce over 16 states
        y += __shfl_xor(y, 1, 16);
        y += __shfl_xor(y, 2, 16);
        y += __shfl_xor(y, 4, 16);
        y += __shfl_xor(y, 8, 16);
        if (s == 0) {
            float zv = xz[row * (2 * D_INNER) + D_INNER + d];
            float sz = zv / (1.0f + __expf(-zv));
            yact[row * D_INNER + d] = (y + dp * u_v) * sz;  // gate, fused
        }
    }
}

extern "C" void kernel_launch(void* const* d_in, const int* in_sizes, int n_in,
                              void* d_out, int out_size, void* d_ws, size_t ws_size,
                              hipStream_t stream)
{
    const float* x        = (const float*)d_in[0];
    const float* in_w     = (const float*)d_in[1];
    const float* conv_w   = (const float*)d_in[2];
    const float* conv_b   = (const float*)d_in[3];
    const float* xproj_w  = (const float*)d_in[4];
    const float* dtproj_w = (const float*)d_in[5];
    const float* dtproj_b = (const float*)d_in[6];
    const float* A_log    = (const float*)d_in[7];
    const float* Dp       = (const float*)d_in[8];
    const float* out_w    = (const float*)d_in[9];

    char* ws = (char*)d_ws;
    size_t off = 0;
    float* xz   = (float*)(ws + off); off += (size_t)MROWS * 2 * D_INNER * 4;  // 128 MB
    float* xbc  = (float*)(ws + off); off += (size_t)MROWS * D_INNER * 4;      //  64 MB
    float* proj = (float*)(ws + off); off += (size_t)MROWS * PROJW * 4;        //   3 MB
    float* dtm  = (float*)(ws + off); off += (size_t)MROWS * D_INNER * 4;      //  64 MB
    float* yact = (float*)(ws + off);                                          //  64 MB

    dim3 blk(256);

    // 1) xz = x @ in_proj_w        (8192 x 4096, K=1024)
    gemm_bf16_wmma<0><<<dim3((2 * D_INNER) / BN, MROWS / BM), blk, 0, stream>>>(
        x, D_MODEL, in_w, 2 * D_INNER, nullptr, xz, 2 * D_INNER,
        MROWS, 2 * D_INNER, D_MODEL);

    // 2) depthwise causal conv + SiLU on xb half
    conv_silu_kernel<<<(BATCH * SEQ * D_INNER) / (256 * 4), blk, 0, stream>>>(
        xz, conv_w, conv_b, xbc);

    // 3) proj = xbc @ x_proj_w     (8192 x 96, K=2048)
    gemm_bf16_wmma<0><<<dim3((PROJW + BN - 1) / BN, MROWS / BM), blk, 0, stream>>>(
        xbc, D_INNER, xproj_w, PROJW, nullptr, proj, PROJW,
        MROWS, PROJW, D_INNER);

    // 4) dt = softplus(proj[:, :64] @ dt_proj_w + dt_proj_b)  (8192 x 2048, K=64)
    gemm_bf16_wmma<1><<<dim3(D_INNER / BN, MROWS / BM), blk, 0, stream>>>(
        proj, PROJW, dtproj_w, D_INNER, dtproj_b, dtm, D_INNER,
        MROWS, D_INNER, DT_RANK);

    // 5) selective scan + gating fused
    scan_kernel<<<dim3(D_INNER / 16, BATCH), blk, 0, stream>>>(
        dtm, xbc, proj, xz, A_log, Dp, yact);

    // 6) out = yact @ out_proj_w   (8192 x 1024, K=2048)
    gemm_bf16_wmma<0><<<dim3(D_MODEL / BN, MROWS / BM), blk, 0, stream>>>(
        yact, D_INNER, out_w, D_MODEL, nullptr, (float*)d_out, D_MODEL,
        MROWS, D_MODEL, D_INNER);
}
